// MultiHeadAttention_16904991277397
// MI455X (gfx1250) — matrix operand order repaired and checked by reading, not compiled, not run on hardware
//
#include <hip/hip_runtime.h>

#define B_   2
#define L_   2048
#define D_   1024
#define H_   16
#define DK_  64
#define LOG2E 1.44269504088896340736f

typedef __attribute__((ext_vector_type(16))) __bf16 v16bf;
typedef __attribute__((ext_vector_type(8)))  __bf16 v8bf;
typedef __attribute__((ext_vector_type(8)))  float  v8f;

// ---------- helpers ----------
__device__ __forceinline__ unsigned short cvt1(float f) {   // HW RNE f32->bf16
  __bf16 h = (__bf16)f;
  return __builtin_bit_cast(unsigned short, h);
}
__device__ __forceinline__ __bf16 f2bf(float f) { return (__bf16)f; }

// A-matrix 16x32 bf16 fragment: lane m = lane&15,
// k = (lane>>4)*16 + e.  rowp points at [row][k_base].
__device__ __forceinline__ v16bf a_frag_bf16(const __bf16* rowp, int hf) {
  v8bf lo = *(const v8bf*)(rowp + hf * 16);
  v8bf hi = *(const v8bf*)(rowp + hf * 16 + 8);
  v16bf r;
#pragma unroll
  for (int i = 0; i < 8; i++) { r[i] = lo[i]; r[i + 8] = hi[i]; }
  return r;
}
// B-matrix 32x16 bf16 fragment: lane n = lane&15, k = (lane>>4)*16 + e
// -> 16 contiguous bf16 per lane (one 32B load).

#define WMMA_BF16(Af, Bf, Cf) \
  __builtin_amdgcn_wmma_f32_16x16x32_bf16(false, (Af), false, (Bf), (short)0, (Cf), false, false)

// ---------------------------------------------------------------------
// Shared ping-pong software-pipelined GEMM K-loop: 32(M)x64(N) per wave,
// two disjoint fragment register sets -> no rotation copies, loads for
// stage i+1 issue while stage i's 8 WMMAs execute.
// ---------------------------------------------------------------------
__device__ __forceinline__ void gemm_loop(
    const __bf16* __restrict__ arow0, const __bf16* __restrict__ arow1,
    const __bf16* __restrict__ wr0, const __bf16* __restrict__ wr1,
    const __bf16* __restrict__ wr2, const __bf16* __restrict__ wr3,
    int hf, v8f acc[8])
{
  v16bf a0 = a_frag_bf16(arow0, hf);
  v16bf a1 = a_frag_bf16(arow1, hf);
  v16bf b0 = *(const v16bf*)(wr0);
  v16bf b1 = *(const v16bf*)(wr1);
  v16bf b2 = *(const v16bf*)(wr2);
  v16bf b3 = *(const v16bf*)(wr3);

#pragma unroll 1
  for (int kb = 0; kb < D_; kb += 64) {
    const int k1 = kb + 32;
    v16bf c0 = a_frag_bf16(arow0 + k1, hf);
    v16bf c1 = a_frag_bf16(arow1 + k1, hf);
    v16bf d0 = *(const v16bf*)(wr0 + k1);
    v16bf d1 = *(const v16bf*)(wr1 + k1);
    v16bf d2 = *(const v16bf*)(wr2 + k1);
    v16bf d3 = *(const v16bf*)(wr3 + k1);

    acc[0] = WMMA_BF16(a0, b0, acc[0]);
    acc[4] = WMMA_BF16(a1, b0, acc[4]);
    acc[1] = WMMA_BF16(a0, b1, acc[1]);
    acc[5] = WMMA_BF16(a1, b1, acc[5]);
    acc[2] = WMMA_BF16(a0, b2, acc[2]);
    acc[6] = WMMA_BF16(a1, b2, acc[6]);
    acc[3] = WMMA_BF16(a0, b3, acc[3]);
    acc[7] = WMMA_BF16(a1, b3, acc[7]);

    const int k2 = (kb + 64) & (D_ - 1);      // wraps on last iter (in-bounds)
    a0 = a_frag_bf16(arow0 + k2, hf);
    a1 = a_frag_bf16(arow1 + k2, hf);
    b0 = *(const v16bf*)(wr0 + k2);
    b1 = *(const v16bf*)(wr1 + k2);
    b2 = *(const v16bf*)(wr2 + k2);
    b3 = *(const v16bf*)(wr3 + k2);

    acc[0] = WMMA_BF16(c0, d0, acc[0]);
    acc[4] = WMMA_BF16(c1, d0, acc[4]);
    acc[1] = WMMA_BF16(c0, d1, acc[1]);
    acc[5] = WMMA_BF16(c1, d1, acc[5]);
    acc[2] = WMMA_BF16(c0, d2, acc[2]);
    acc[6] = WMMA_BF16(c1, d2, acc[6]);
    acc[3] = WMMA_BF16(c0, d3, acc[3]);
    acc[7] = WMMA_BF16(c1, d3, acc[7]);
  }
}

// =====================================================================
// Kernel 0: bulk fp32 -> bf16 conversion (bandwidth-bound, one pass).
// =====================================================================
__global__ __launch_bounds__(256) void cvt_kernel(
    const float* __restrict__ src, unsigned short* __restrict__ dst, int n4)
{
  int i = blockIdx.x * 256 + threadIdx.x;
  if (i < n4) {
    float4 v = ((const float4*)src)[i];
    unsigned long long p =
        (unsigned long long)cvt1(v.x) |
        ((unsigned long long)cvt1(v.y) << 16) |
        ((unsigned long long)cvt1(v.z) << 32) |
        ((unsigned long long)cvt1(v.w) << 48);
    ((unsigned long long*)dst)[i] = p;
  }
}

// =====================================================================
// Kernel 1: QKV projections.  C = A(bf16) @ W^T(bf16) + bias, out bf16.
// grid = (M/256, N/64, 3).  Wave computes a 32x64 tile.
// Q,K stored [h][b][l][64]; V stored transposed [h][b][64][l].
// =====================================================================
__global__ __launch_bounds__(256) void qkv_gemm_kernel(
    const unsigned short* __restrict__ qa, const unsigned short* __restrict__ ka,
    const unsigned short* __restrict__ va,
    const unsigned short* __restrict__ wqb, const float* __restrict__ bq,
    const unsigned short* __restrict__ wkb, const float* __restrict__ bk,
    const unsigned short* __restrict__ wvb, const float* __restrict__ bv,
    unsigned short* __restrict__ qh_u, unsigned short* __restrict__ kh_u,
    unsigned short* __restrict__ vt_u)
{
  const int lane = threadIdx.x & 31, wave = threadIdx.x >> 5;
  const int hf = lane >> 4, l16 = lane & 15;
  const int m0 = blockIdx.x * 256 + wave * 32;
  const int n0 = blockIdx.y * 64;
  const int sel = blockIdx.z;

  const __bf16* A    = (const __bf16*)((sel == 0) ? qa  : (sel == 1) ? ka  : va);
  const __bf16* W    = (const __bf16*)((sel == 0) ? wqb : (sel == 1) ? wkb : wvb);
  const float*  bias = (sel == 0) ? bq : (sel == 1) ? bk : bv;

  v8f acc[8] = {};
  gemm_loop(A + (size_t)(m0 + l16) * D_,
            A + (size_t)(m0 + 16 + l16) * D_,
            W + (size_t)(n0 +  0 + l16) * D_ + hf * 16,
            W + (size_t)(n0 + 16 + l16) * D_ + hf * 16,
            W + (size_t)(n0 + 32 + l16) * D_ + hf * 16,
            W + (size_t)(n0 + 48 + l16) * D_ + hf * 16,
            hf, acc);

#pragma unroll
  for (int ms = 0; ms < 2; ms++)
#pragma unroll
    for (int nt = 0; nt < 4; nt++)
#pragma unroll
      for (int r = 0; r < 8; r++) {
        int mm = m0 + ms * 16 + hf * 8 + r;   // D layout: m = half*8 + r
        int nn = n0 + nt * 16 + l16;          //            n = lane&15
        float v = acc[ms * 4 + nt][r] + bias[nn];
        int b = mm >> 11, l = mm & (L_ - 1);
        int h = nn >> 6,  d = nn & 63;
        int hb = h * B_ + b;
        unsigned short o = cvt1(v);
        if (sel == 2) vt_u[((size_t)hb * 64 + d) * L_ + l] = o;     // V transposed
        else {
          unsigned short* dst = (sel == 0) ? qh_u : kh_u;
          dst[((size_t)hb * L_ + l) * 64 + d] = o;
        }
      }
}

// =====================================================================
// Kernel 2: attention, two-pass flash per 128-row query block.
// grid = (L/128, H*B).  Each wave: 16 query rows.  KV tile = 64.
// Writes exact softmax probs (fp32) to attn_out and bf16 ctx.
// =====================================================================
__global__ __launch_bounds__(256) void attn_kernel(
    const unsigned short* __restrict__ qh_u, const unsigned short* __restrict__ kh_u,
    const unsigned short* __restrict__ vt_u,
    unsigned short* __restrict__ ctx_u, float* __restrict__ attn_out)
{
  __shared__ __attribute__((aligned(32))) __bf16 s_p[8][16][64];

  const int lane = threadIdx.x & 31, wave = threadIdx.x >> 5;
  const int hf = lane >> 4, l16 = lane & 15;
  const int hb = blockIdx.y;
  const int q0 = blockIdx.x * 128 + wave * 16;

  const __bf16* qh = (const __bf16*)qh_u;
  const __bf16* kh = (const __bf16*)kh_u + (size_t)hb * L_ * 64;
  const __bf16* vt = (const __bf16*)vt_u + (size_t)hb * 64 * L_;

  const __bf16* qrow = qh + ((size_t)hb * L_ + q0 + l16) * 64;
  v16bf aq0 = a_frag_bf16(qrow,      hf);   // dk 0..31
  v16bf aq1 = a_frag_bf16(qrow + 32, hf);   // dk 32..63

  float rm[8], rl[8];
#pragma unroll
  for (int r = 0; r < 8; r++) { rm[r] = -1e30f; rl[r] = 0.f; }

  // ---------------- pass 1: row max / sum ----------------
#pragma unroll 1
  for (int t = 0; t < L_; t += 64) {
    v8f sacc[4] = {};
#pragma unroll
    for (int nt = 0; nt < 4; nt++) {
      const __bf16* kr = kh + (size_t)(t + nt * 16 + l16) * 64;
      sacc[nt] = WMMA_BF16(aq0, *(const v16bf*)(kr + hf * 16),      sacc[nt]);
      sacc[nt] = WMMA_BF16(aq1, *(const v16bf*)(kr + 32 + hf * 16), sacc[nt]);
    }
#pragma unroll
    for (int r = 0; r < 8; r++) {
      float tm = -1e30f;
#pragma unroll
      for (int nt = 0; nt < 4; nt++) tm = fmaxf(tm, sacc[nt][r]);
      tm *= 0.125f;                                        // 1/sqrt(64), >0
#pragma unroll
      for (int off = 1; off < 16; off <<= 1) tm = fmaxf(tm, __shfl_xor(tm, off, 32));
      float newm = fmaxf(rm[r], tm);
      float ps = 0.f;
#pragma unroll
      for (int nt = 0; nt < 4; nt++)
        ps += __builtin_amdgcn_exp2f((sacc[nt][r] * 0.125f - newm) * LOG2E);
#pragma unroll
      for (int off = 1; off < 16; off <<= 1) ps += __shfl_xor(ps, off, 32);
      rl[r] = rl[r] * __builtin_amdgcn_exp2f((rm[r] - newm) * LOG2E) + ps;
      rm[r] = newm;
    }
  }
  float ril[8];
#pragma unroll
  for (int r = 0; r < 8; r++) ril[r] = 1.0f / rl[r];

  // ---------------- pass 2: probs out + ctx = P @ V ----------------
  v8f cacc[4] = {};
#pragma unroll 1
  for (int t = 0; t < L_; t += 64) {
    v8f sacc[4] = {};
#pragma unroll
    for (int nt = 0; nt < 4; nt++) {
      const __bf16* kr = kh + (size_t)(t + nt * 16 + l16) * 64;
      sacc[nt] = WMMA_BF16(aq0, *(const v16bf*)(kr + hf * 16),      sacc[nt]);
      sacc[nt] = WMMA_BF16(aq1, *(const v16bf*)(kr + 32 + hf * 16), sacc[nt]);
    }
#pragma unroll
    for (int nt = 0; nt < 4; nt++)
#pragma unroll
      for (int r = 0; r < 8; r++) {
        float p = __builtin_amdgcn_exp2f((sacc[nt][r] * 0.125f - rm[r]) * LOG2E) * ril[r];
        attn_out[((size_t)hb * L_ + (q0 + hf * 8 + r)) * L_ + t + nt * 16 + l16] = p;
        s_p[wave][hf * 8 + r][nt * 16 + l16] = f2bf(p);
      }
    asm volatile("s_wait_dscnt 0" ::: "memory");   // same-wave DS order fence
#pragma unroll
    for (int c2 = 0; c2 < 2; c2++) {
      v16bf ap = a_frag_bf16(&s_p[wave][l16][c2 * 32], hf);
#pragma unroll
      for (int ct = 0; ct < 4; ct++) {
        const __bf16* vp = vt + (size_t)(ct * 16 + l16) * L_ + t + c2 * 32 + hf * 16;
        cacc[ct] = WMMA_BF16(ap, *(const v16bf*)vp, cacc[ct]);
      }
    }
  }

  const int h = hb / B_, b = hb % B_;
#pragma unroll
  for (int ct = 0; ct < 4; ct++)
#pragma unroll
    for (int r = 0; r < 8; r++) {
      int row = q0 + hf * 8 + r;
      ctx_u[((size_t)b * L_ + row) * (H_ * DK_) + h * 64 + ct * 16 + l16] =
          cvt1(cacc[ct][r]);     // already normalized by 1/l
    }
}

// =====================================================================
// Kernel 3: out_pre = ctx(bf16) @ fc_w^T(bf16) + fc_b + residual (fp32)
// grid = (M/256, N/64); same pipelined 32x64 wave tile.
// =====================================================================
__global__ __launch_bounds__(256) void out_gemm_kernel(
    const unsigned short* __restrict__ ctx_u, const unsigned short* __restrict__ fcw_u,
    const float* __restrict__ fcb, const float* __restrict__ resid,
    float* __restrict__ out_pre)
{
  const int lane = threadIdx.x & 31, wave = threadIdx.x >> 5;
  const int hf = lane >> 4, l16 = lane & 15;
  const int m0 = blockIdx.x * 256 + wave * 32;
  const int n0 = blockIdx.y * 64;

  const __bf16* A = (const __bf16*)ctx_u;
  const __bf16* W = (const __bf16*)fcw_u;

  v8f acc[8] = {};
  gemm_loop(A + (size_t)(m0 + l16) * D_,
            A + (size_t)(m0 + 16 + l16) * D_,
            W + (size_t)(n0 +  0 + l16) * D_ + hf * 16,
            W + (size_t)(n0 + 16 + l16) * D_ + hf * 16,
            W + (size_t)(n0 + 32 + l16) * D_ + hf * 16,
            W + (size_t)(n0 + 48 + l16) * D_ + hf * 16,
            hf, acc);

#pragma unroll
  for (int ms = 0; ms < 2; ms++)
#pragma unroll
    for (int nt = 0; nt < 4; nt++)
#pragma unroll
      for (int r = 0; r < 8; r++) {
        int mm = m0 + ms * 16 + hf * 8 + r;
        int nn = n0 + nt * 16 + l16;
        out_pre[(size_t)mm * D_ + nn] =
            acc[ms * 4 + nt][r] + fcb[nn] + resid[(size_t)mm * D_ + nn];
      }
}

// =====================================================================
// Kernel 4: LayerNorm over D=1024, one block per row.
// =====================================================================
__global__ __launch_bounds__(256) void ln_kernel(
    const float* __restrict__ pre, const float* __restrict__ g,
    const float* __restrict__ bb, float* __restrict__ out)
{
  const int row = blockIdx.x;
  const int lane = threadIdx.x & 31, wave = threadIdx.x >> 5;
  const float* p = pre + (size_t)row * D_;

  float s = 0.f, s2 = 0.f;
#pragma unroll
  for (int i = threadIdx.x; i < D_; i += 256) { float v = p[i]; s += v; s2 += v * v; }
#pragma unroll
  for (int off = 1; off < 32; off <<= 1) {
    s  += __shfl_xor(s,  off, 32);
    s2 += __shfl_xor(s2, off, 32);
  }
  __shared__ float ws1[8], ws2[8];
  if (lane == 0) { ws1[wave] = s; ws2[wave] = s2; }
  __syncthreads();
  float ts = 0.f, ts2 = 0.f;
#pragma unroll
  for (int i = 0; i < 8; i++) { ts += ws1[i]; ts2 += ws2[i]; }
  float mu  = ts * (1.0f / D_);
  float var = ts2 * (1.0f / D_) - mu * mu;
  float rs  = rsqrtf(var + 1e-5f);
#pragma unroll
  for (int i = threadIdx.x; i < D_; i += 256)
    out[(size_t)row * D_ + i] = (p[i] - mu) * rs * g[i] + bb[i];
}

// =====================================================================
extern "C" void kernel_launch(void* const* d_in, const int* in_sizes, int n_in,
                              void* d_out, int out_size, void* d_ws, size_t ws_size,
                              hipStream_t stream)
{
  (void)in_sizes; (void)n_in; (void)out_size; (void)ws_size;
  const float* q    = (const float*)d_in[0];
  const float* k    = (const float*)d_in[1];
  const float* v    = (const float*)d_in[2];
  const float* wq_w = (const float*)d_in[3];
  const float* wq_b = (const float*)d_in[4];
  const float* wk_w = (const float*)d_in[5];
  const float* wk_b = (const float*)d_in[6];
  const float* wv_w = (const float*)d_in[7];
  const float* wv_b = (const float*)d_in[8];
  const float* fc_w = (const float*)d_in[9];
  const float* fc_b = (const float*)d_in[10];
  const float* ln_g = (const float*)d_in[11];
  const float* ln_b = (const float*)d_in[12];

  char* ws = (char*)d_ws;
  const size_t MB = 1024ull * 1024ull;
  // bf16 copies of activations and weights (one-time conversion)
  unsigned short* qbf  = (unsigned short*)(ws + 0 * MB);   // 8 MB  [b*l][1024]
  unsigned short* kbf  = (unsigned short*)(ws + 8 * MB);   // 8 MB
  unsigned short* vbf  = (unsigned short*)(ws + 16 * MB);  // 8 MB
  unsigned short* wqb  = (unsigned short*)(ws + 24 * MB);  // 2 MB  [1024][1024]
  unsigned short* wkb  = (unsigned short*)(ws + 26 * MB);  // 2 MB
  unsigned short* wvb  = (unsigned short*)(ws + 28 * MB);  // 2 MB
  unsigned short* fcwb = (unsigned short*)(ws + 30 * MB);  // 2 MB
  // intermediates
  unsigned short* qh   = (unsigned short*)(ws + 32 * MB);  // 8 MB  [h][b][l][64]
  unsigned short* kh   = (unsigned short*)(ws + 40 * MB);  // 8 MB  [h][b][l][64]
  unsigned short* vt   = (unsigned short*)(ws + 48 * MB);  // 8 MB  [h][b][64][l]
  unsigned short* ctx  = (unsigned short*)(ws + 56 * MB);  // 8 MB  [b][l][1024]
  float* out_pre       = (float*)(ws + 64 * MB);           // 16 MB [b*l][1024]

  float* outp     = (float*)d_out;
  float* attn_out = outp + (size_t)B_ * L_ * D_;           // probs [h][b][l][t]

  const int nAct4 = (B_ * L_ * D_) / 4;     // 1,048,576
  const int nW4   = (D_ * D_) / 4;          //   262,144
  cvt_kernel<<<(nAct4 + 255) / 256, 256, 0, stream>>>(q,    qbf,  nAct4);
  cvt_kernel<<<(nAct4 + 255) / 256, 256, 0, stream>>>(k,    kbf,  nAct4);
  cvt_kernel<<<(nAct4 + 255) / 256, 256, 0, stream>>>(v,    vbf,  nAct4);
  cvt_kernel<<<(nW4   + 255) / 256, 256, 0, stream>>>(wq_w, wqb,  nW4);
  cvt_kernel<<<(nW4   + 255) / 256, 256, 0, stream>>>(wk_w, wkb,  nW4);
  cvt_kernel<<<(nW4   + 255) / 256, 256, 0, stream>>>(wv_w, wvb,  nW4);
  cvt_kernel<<<(nW4   + 255) / 256, 256, 0, stream>>>(fc_w, fcwb, nW4);

  qkv_gemm_kernel<<<dim3(16, 16, 3), 256, 0, stream>>>(
      qbf, kbf, vbf, wqb, wq_b, wkb, wk_b, wvb, wv_b, qh, kh, vt);
  attn_kernel<<<dim3(L_ / 128, H_ * B_), 256, 0, stream>>>(
      qh, kh, vt, ctx, attn_out);
  out_gemm_kernel<<<dim3(16, 16), 256, 0, stream>>>(
      ctx, fcwb, fc_b, q, out_pre);
  ln_kernel<<<B_ * L_, 256, 0, stream>>>(out_pre, ln_g, ln_b, outp);
}
